// BRITSCell_9740985827796
// MI455X (gfx1250) — matrix operand order repaired and checked by reading, not compiled, not run on hardware
//
#include <hip/hip_runtime.h>
#include <hip/hip_bf16.h>

#define Bsz 8192
#define Tsz 256
#define Hsz 128
#define BM  64          // batch rows per block
#define NTHREADS 128    // 4 waves, one 16-row tile each

typedef __bf16 v16bf __attribute__((ext_vector_type(16)));
typedef __bf16 v8bf  __attribute__((ext_vector_type(8)));
typedef float  v8f   __attribute__((ext_vector_type(8)));

__device__ __forceinline__ float sigmoid_(float x) {
    return 1.0f / (1.0f + __expf(-x));
}
__device__ __forceinline__ float tanh_(float x) {
    x = fminf(fmaxf(x, -15.0f), 15.0f);
    float e = __expf(2.0f * x);
    return (e - 1.0f) / (e + 1.0f);
}

__device__ __forceinline__ v8f wmma_bf16(v16bf a, v16bf b, v8f c) {
    return __builtin_amdgcn_wmma_f32_16x16x32_bf16(
        /*neg_a=*/false, a, /*neg_b=*/false, b,
        /*c_mod=*/(short)0, c, /*reuse_a=*/false, /*reuse_b=*/false);
}

// A fragment (16x32, bf16) from bf16 LDS tile (row-major, stride Hsz).
// CDNA5 16-bit A layout: lanes 0-15 -> K[k0+0..7] then K[k0+16..23];
//                        lanes 16-31 -> K[k0+8..15] then K[k0+24..31].
__device__ __forceinline__ v16bf load_afrag_bf16(const __bf16* base, int row0, int k0, int lane) {
    int r  = row0 + (lane & 15);
    int hi = (lane >> 4) & 1;
    const __bf16* p = base + r * Hsz + k0 + hi * 8;
    union { v16bf w; v8bf h[2]; } u;
    u.h[0] = *(const v8bf*)p;         // 16B ds_load
    u.h[1] = *(const v8bf*)(p + 16);  // 16B ds_load
    return u.w;
}

// Same A layout but sourced from f32 LDS (h state), converted to bf16.
__device__ __forceinline__ v16bf load_afrag_f32(const float* base, int row0, int k0, int lane) {
    int r  = row0 + (lane & 15);
    int hi = (lane >> 4) & 1;
    const float* p = base + r * Hsz + k0 + hi * 8;
    union { v16bf w; __bf16 e[16]; } u;
#pragma unroll
    for (int i = 0; i < 8; ++i) {
        u.e[i]     = (__bf16)p[i];
        u.e[8 + i] = (__bf16)p[16 + i];
    }
    return u.w;
}

// B fragment (32x16): B[k][n] = W[(j0+n)*Hsz + k] with W row-major bf16.
// Lanes 0-15 hold column n=lane with K[k0+0..7] then K[k0+16..23]; lanes 16-31 K[k0+8..15] then K[k0+24..31].
// Two 16-byte LDS loads per lane.
__device__ __forceinline__ v16bf load_bfrag(const __bf16* W, int j0, int k0, int lane) {
    int n  = lane & 15;
    int hi = (lane >> 4) & 1;
    const __bf16* p = W + (j0 + n) * Hsz + k0 + hi * 8;
    union { v16bf w; v8bf h[2]; } u;
    u.h[0] = *(const v8bf*)p;
    u.h[1] = *(const v8bf*)(p + 16);
    return u.w;
}

__global__ __launch_bounds__(NTHREADS)
void brits_gru_kernel(const float* __restrict__ x_seq, const float* __restrict__ m_seq,
                      const float* __restrict__ Wc_w,  const float* __restrict__ Wc_b,
                      const float* __restrict__ Wx_w,  const float* __restrict__ Wx_b,
                      const float* __restrict__ W_ih,  const float* __restrict__ W_hh,
                      const float* __restrict__ b_ih,  const float* __restrict__ b_hh,
                      const float* __restrict__ out_w, const float* __restrict__ out_b,
                      float* __restrict__ preds, float* __restrict__ comps)
{
    extern __shared__ char smem[];
    __bf16* sWih = (__bf16*)smem;            // 384*128 bf16 (first H cols of W_ih)
    __bf16* sWhh = sWih + 384 * Hsz;         // 384*128 bf16
    __bf16* sF16 = sWhh + 384 * Hsz;         // BM*128 bf16 (feat)
    float*  sHf  = (float*)(sF16 + BM * Hsz);// BM*128 f32 (h state)
    float*  sBih = sHf + BM * Hsz;           // 384
    float*  sBhh = sBih + 384;               // 384
    float*  sWm  = sBhh + 384;               // 384 (mask column of W_ih)
    float*  sWcw = sWm + 384;                // 128
    float*  sWxw = sWcw + Hsz;               // 128
    float*  sWxb = sWxw + Hsz;               // 128
    float*  sOw  = sWxb + Hsz;               // 128
    float*  sM   = sOw + Hsz;                // BM (m_t per row)

    const int tid  = threadIdx.x;
    const int lane = tid & 31;
    const int wave = tid >> 5;
    const int wrow = wave * 16;
    const int b0   = blockIdx.x * BM;

    // ---- one-time: load + bf16-convert weights into LDS, zero h ----
    for (int i = tid; i < 384 * Hsz; i += NTHREADS) {
        int j = i >> 7, k = i & 127;
        sWih[i] = (__bf16)W_ih[j * (Hsz + 1) + k];
        sWhh[i] = (__bf16)W_hh[i];
    }
    for (int i = tid; i < 384; i += NTHREADS) {
        sBih[i] = b_ih[i];
        sBhh[i] = b_hh[i];
        sWm[i]  = W_ih[i * (Hsz + 1) + Hsz];
    }
    if (tid < Hsz) {
        sWcw[tid] = Wc_w[tid];
        sWxw[tid] = Wx_w[tid];
        sWxb[tid] = Wx_b[tid];
        sOw[tid]  = out_w[tid];
    }
    for (int i = tid; i < BM * Hsz; i += NTHREADS) sHf[i] = 0.0f;
    const float wcb  = Wc_b[0];
    const float outb = out_b[0];
    __syncthreads();

    // Phase-A work assignment: 2 threads per batch row, each covers 64 K's.
    // Note rows [16w,16w+16) map exactly to wave w's threads, so all per-step
    // data flow (h, feat, m) is wave-local: no per-step barriers required.
    const int rowA   = tid >> 1;
    const int halfA  = tid & 1;
    const int kbaseA = halfA * 64;
    const int bA     = b0 + rowA;
    const int hi     = (lane >> 4) & 1;

    for (int t = 0; t < Tsz; ++t) {
        // ================= Phase A: c_t, x_imp, feat =================
        float dot = 0.0f;
        const float4* hp = (const float4*)(sHf + rowA * Hsz + kbaseA);
        const float4* wp = (const float4*)(sWcw + kbaseA);
#pragma unroll 4
        for (int k = 0; k < 16; ++k) {
            float4 a = hp[k], w = wp[k];
            dot = fmaf(a.x, w.x, fmaf(a.y, w.y, fmaf(a.z, w.z, fmaf(a.w, w.w, dot))));
        }
        dot += __shfl_xor(dot, 1, 32);     // combine the two K-halves
        float c = tanh_(dot + wcb);
        float m = m_seq[(size_t)bA * Tsz + t];
        float x = x_seq[(size_t)bA * Tsz + t];
        float xi = m * x + (1.0f - m) * c;
        if (halfA == 0) {
            comps[(size_t)bA * Tsz + t] = c;
            sM[rowA] = m;
        }
        for (int k8 = kbaseA; k8 < kbaseA + 64; k8 += 8) {
            v8bf pk;
#pragma unroll
            for (int i = 0; i < 8; ++i)
                pk[i] = (__bf16)fmaxf(fmaf(xi, sWxw[k8 + i], sWxb[k8 + i]), 0.0f);
            *(v8bf*)(sF16 + rowA * Hsz + k8) = pk;  // 16B ds_store
        }

        // ================= Phase B: fused gate GEMMs (WMMA) =================
        v16bf fA[4], hA[4];
#pragma unroll
        for (int kk = 0; kk < 4; ++kk) {
            fA[kk] = load_afrag_bf16(sF16, wrow, kk * 32, lane);
            hA[kk] = load_afrag_f32 (sHf,  wrow, kk * 32, lane);
        }
        float mv[8];
#pragma unroll
        for (int v = 0; v < 8; ++v) mv[v] = sM[wrow + 8 * hi + v];
        float pred[8];
#pragma unroll
        for (int v = 0; v < 8; ++v) pred[v] = 0.0f;

        for (int j = 0; j < 8; ++j) {           // 8 column tiles of 16
            v8f accR = {}, accZ = {}, accN1 = {}, accN2 = {};
#pragma unroll
            for (int kk = 0; kk < 4; ++kk) {
                int k0 = kk * 32;
                accR  = wmma_bf16(fA[kk], load_bfrag(sWih,       j * 16, k0, lane), accR);
                accR  = wmma_bf16(hA[kk], load_bfrag(sWhh,       j * 16, k0, lane), accR);
                accZ  = wmma_bf16(fA[kk], load_bfrag(sWih, 128 + j * 16, k0, lane), accZ);
                accZ  = wmma_bf16(hA[kk], load_bfrag(sWhh, 128 + j * 16, k0, lane), accZ);
                accN1 = wmma_bf16(fA[kk], load_bfrag(sWih, 256 + j * 16, k0, lane), accN1);
                accN2 = wmma_bf16(hA[kk], load_bfrag(sWhh, 256 + j * 16, k0, lane), accN2);
            }
            int col = j * 16 + (lane & 15);
            float bir = sBih[col],       bhr = sBhh[col],       wmr = sWm[col];
            float biz = sBih[128 + col], bhz = sBhh[128 + col], wmz = sWm[128 + col];
            float bin = sBih[256 + col], bhn = sBhh[256 + col], wmn = sWm[256 + col];
            float ow  = sOw[col];
#pragma unroll
            for (int v = 0; v < 8; ++v) {
                int rloc = wrow + 8 * hi + v;          // D layout: m = v + 8*hi
                float r  = sigmoid_(accR[v] + bir + bhr + mv[v] * wmr);
                float z  = sigmoid_(accZ[v] + biz + bhz + mv[v] * wmz);
                float n  = tanh_(accN1[v] + bin + mv[v] * wmn + r * (accN2[v] + bhn));
                float ho = sHf[rloc * Hsz + col];
                float hn = (1.0f - z) * n + z * ho;
                sHf[rloc * Hsz + col] = hn;
                pred[v] = fmaf(hn, ow, pred[v]);
            }
        }
        // reduce pred over the 16 columns held across each 16-lane half
#pragma unroll
        for (int v = 0; v < 8; ++v) {
            float p = pred[v];
            p += __shfl_xor(p, 1, 32);
            p += __shfl_xor(p, 2, 32);
            p += __shfl_xor(p, 4, 32);
            p += __shfl_xor(p, 8, 32);
            pred[v] = p;
        }
        if ((lane & 15) == 0) {
#pragma unroll
            for (int v = 0; v < 8; ++v) {
                int bg = b0 + wrow + 8 * hi + v;
                preds[(size_t)bg * Tsz + t] = pred[v] + outb;
            }
        }
        // no __syncthreads: all per-step LDS traffic is wave-local;
        // weights are read-only after the initial barrier.
    }
}

extern "C" void kernel_launch(void* const* d_in, const int* in_sizes, int n_in,
                              void* d_out, int out_size, void* d_ws, size_t ws_size,
                              hipStream_t stream) {
    (void)in_sizes; (void)n_in; (void)out_size; (void)d_ws; (void)ws_size;
    const float* x_seq = (const float*)d_in[0];
    const float* m_seq = (const float*)d_in[1];
    const float* Wc_w  = (const float*)d_in[2];
    const float* Wc_b  = (const float*)d_in[3];
    const float* Wx_w  = (const float*)d_in[4];
    const float* Wx_b  = (const float*)d_in[5];
    const float* W_ih  = (const float*)d_in[6];
    const float* W_hh  = (const float*)d_in[7];
    const float* b_ih  = (const float*)d_in[8];
    const float* b_hh  = (const float*)d_in[9];
    const float* out_w = (const float*)d_in[10];
    const float* out_b = (const float*)d_in[11];

    float* preds = (float*)d_out;
    float* comps = preds + (size_t)Bsz * Tsz;

    const size_t shmem =
        (size_t)(384 * Hsz) * 2 * sizeof(__bf16)      // sWih + sWhh
      + (size_t)(BM * Hsz) * sizeof(__bf16)           // sF16
      + (size_t)(BM * Hsz) * sizeof(float)            // sHf
      + (size_t)(384 * 3) * sizeof(float)             // sBih, sBhh, sWm
      + (size_t)(Hsz * 4) * sizeof(float)             // sWcw, sWxw, sWxb, sOw
      + (size_t)BM * sizeof(float);                   // sM

    hipFuncSetAttribute((const void*)brits_gru_kernel,
                        hipFuncAttributeMaxDynamicSharedMemorySize, (int)shmem);

    brits_gru_kernel<<<Bsz / BM, NTHREADS, shmem, stream>>>(
        x_seq, m_seq, Wc_w, Wc_b, Wx_w, Wx_b,
        W_ih, W_hh, b_ih, b_hh, out_w, out_b, preds, comps);
}
